// GCN_AQ_80573586473109
// MI455X (gfx1250) — matrix operand order repaired and checked by reading, not compiled, not run on hardware
//
#include <hip/hip_runtime.h>
#include <cstdint>

typedef __attribute__((ext_vector_type(16))) _Float16 v16h;
typedef __attribute__((ext_vector_type(8)))  float    v8f;

#define DEVFN __device__ __forceinline__

constexpr int B_ = 16, N_ = 10000, T_ = 24, E_ = 160000, H_ = 32, O_ = 16;
constexpr int M_ = B_ * N_;         // 160000 rows
constexpr int TILES_ = M_ / 16;     // 10000 row tiles

// hardware reciprocal: single v_rcp_f32 (TRANS), avoids IEEE div expansion
DEVFN float frcp(float v) { return __builtin_amdgcn_rcpf(v); }
DEVFN float sigf(float v) { return frcp(1.0f + __expf(-v)); }
DEVFN float tanhfast(float v) {
  v = fminf(fmaxf(v, -15.0f), 15.0f);
  float e = __expf(2.0f * v);           // v_exp_f32
  return 1.0f - 2.0f * frcp(e + 1.0f);  // v_rcp_f32
}
// contiguous K index used here (not AMD's table order) for A element pair p (0..7), half-wave khalf
DEVFN int kmapA(int p, int khalf) {
  return 2 * p + 16 * khalf;
}

// ---------------------------------------------------------------------------
// GRU (24 steps, WMMA 16x16x32 f16 for h @ w_hh^T) fused with xw1 = h @ W1.
// One wave = one 16-row tile; 4 waves / block.
// ---------------------------------------------------------------------------
__global__ __launch_bounds__(128)
void k_gru(const float* __restrict__ x, const float* __restrict__ w_ih,
           const float* __restrict__ w_hh, const float* __restrict__ b_ih,
           const float* __restrict__ b_hh, const float* __restrict__ W1,
           float* __restrict__ xw1) {
  __shared__ float xs[4 * T_ * 16];    // [wave][t][row]
  __shared__ float hst[4 * 16 * 33];   // [wave][row][col] padded stride 33
  const int lane  = threadIdx.x & 31;
  const int wid   = threadIdx.x >> 5;
  const int tile  = blockIdx.x * 4 + wid;
  const int m0    = tile * 16;
  const int kcol  = lane & 15;
  const int khalf = lane >> 4;

  float* xsw = &xs[wid * T_ * 16];
  float* hsw = &hst[wid * 16 * 33];

  // stage the 16x24 x-tile into LDS
  for (int idx = lane; idx < T_ * 16; idx += 32) {
    int row = idx & 15, t = idx >> 4;
    xsw[idx] = x[(size_t)(m0 + row) * T_ + t];
  }

  // per-lane input-gate weights/biases (column = 16*jj + kcol of each gate)
  float wir[2], wiz[2], win[2], bir[2], biz[2], bin_[2], bhr[2], bhz[2], bhn[2];
  #pragma unroll
  for (int jj = 0; jj < 2; ++jj) {
    int kk = 16 * jj + kcol;
    wir[jj] = w_ih[kk];  wiz[jj] = w_ih[H_ + kk];  win[jj]  = w_ih[2 * H_ + kk];
    bir[jj] = b_ih[kk];  biz[jj] = b_ih[H_ + kk];  bin_[jj] = b_ih[2 * H_ + kk];
    bhr[jj] = b_hh[kk];  bhz[jj] = b_hh[H_ + kk];  bhn[jj]  = b_hh[2 * H_ + kk];
  }

  // B tiles (32x16 f16): B_j[k][n] = w_hh[16j+n][k]; lanes0-15 K=0..15, 16-31 K=16..31
  v16h Bw[6];
  #pragma unroll
  for (int j = 0; j < 6; ++j) {
    int g = 16 * j + kcol;
    #pragma unroll
    for (int p = 0; p < 8; ++p) {
      int k = 2 * p + 16 * khalf;
      Bw[j][2 * p]     = (_Float16)w_hh[g * H_ + k];
      Bw[j][2 * p + 1] = (_Float16)w_hh[g * H_ + k + 1];
    }
  }
  // B tiles for xw1 = h @ W1 : B_j[k][n] = W1[k][16j+n]
  v16h Bw1[2];
  #pragma unroll
  for (int j = 0; j < 2; ++j) {
    #pragma unroll
    for (int p = 0; p < 8; ++p) {
      int k = 2 * p + 16 * khalf;
      Bw1[j][2 * p]     = (_Float16)W1[k * H_ + 16 * j + kcol];
      Bw1[j][2 * p + 1] = (_Float16)W1[(k + 1) * H_ + 16 * j + kcol];
    }
  }

  // hidden state in C layout: Hc[jj] covers cols 16*jj..16*jj+15, row = slot+8*khalf
  v8f Hc[2];
  #pragma unroll
  for (int jj = 0; jj < 2; ++jj)
    #pragma unroll
    for (int p = 0; p < 8; ++p) Hc[jj][p] = 0.0f;

  const v8f cz = {0,0,0,0,0,0,0,0};

  for (int t = 0; t < T_; ++t) {
    // C layout -> A layout via LDS transpose
    __syncthreads();
    #pragma unroll
    for (int jj = 0; jj < 2; ++jj)
      #pragma unroll
      for (int p = 0; p < 8; ++p)
        hsw[(p + 8 * khalf) * 33 + 16 * jj + kcol] = Hc[jj][p];
    __syncthreads();
    v16h A;
    #pragma unroll
    for (int p = 0; p < 8; ++p) {
      int k = kmapA(p, khalf);
      A[2 * p]     = (_Float16)hsw[kcol * 33 + k];
      A[2 * p + 1] = (_Float16)hsw[kcol * 33 + k + 1];
    }

    // gh = h @ w_hh^T : six 16x16 tiles (r: 0,1  z: 2,3  n: 4,5)
    v8f G[6];
    #pragma unroll
    for (int j = 0; j < 6; ++j)
      G[j] = __builtin_amdgcn_wmma_f32_16x16x32_f16(false, A, false, Bw[j],
                                                    (short)0, cz, false, false);

    float xv[8];
    #pragma unroll
    for (int p = 0; p < 8; ++p) xv[p] = xsw[t * 16 + p + 8 * khalf];

    #pragma unroll
    for (int jj = 0; jj < 2; ++jj) {
      #pragma unroll
      for (int p = 0; p < 8; ++p) {
        float ghr = G[jj][p]     + bhr[jj];
        float ghz = G[2 + jj][p] + bhz[jj];
        float ghn = G[4 + jj][p] + bhn[jj];
        float r  = sigf(xv[p] * wir[jj] + bir[jj] + ghr);
        float z  = sigf(xv[p] * wiz[jj] + biz[jj] + ghz);
        float nn = tanhfast(xv[p] * win[jj] + bin_[jj] + r * ghn);
        Hc[jj][p] = (1.0f - z) * nn + z * Hc[jj][p];
      }
    }
  }

  // fused xw1 = hT @ W1
  __syncthreads();
  #pragma unroll
  for (int jj = 0; jj < 2; ++jj)
    #pragma unroll
    for (int p = 0; p < 8; ++p)
      hsw[(p + 8 * khalf) * 33 + 16 * jj + kcol] = Hc[jj][p];
  __syncthreads();
  v16h A;
  #pragma unroll
  for (int p = 0; p < 8; ++p) {
    int k = kmapA(p, khalf);
    A[2 * p]     = (_Float16)hsw[kcol * 33 + k];
    A[2 * p + 1] = (_Float16)hsw[kcol * 33 + k + 1];
  }
  v8f XW[2];
  #pragma unroll
  for (int j = 0; j < 2; ++j)
    XW[j] = __builtin_amdgcn_wmma_f32_16x16x32_f16(false, A, false, Bw1[j],
                                                   (short)0, cz, false, false);
  #pragma unroll
  for (int jj = 0; jj < 2; ++jj)
    #pragma unroll
    for (int p = 0; p < 8; ++p)
      xw1[(size_t)(m0 + p + 8 * khalf) * H_ + 16 * jj + kcol] = XW[jj][p];
}

// ---------------------------------------------------------------------------
// xw2 = h1 @ W2 ([160000,32] x [32,16]) — one WMMA per 16-row tile.
// ---------------------------------------------------------------------------
__global__ __launch_bounds__(128)
void k_xw2(const float* __restrict__ h1, const float* __restrict__ W2,
           float* __restrict__ xw2) {
  const int lane  = threadIdx.x & 31;
  const int wid   = threadIdx.x >> 5;
  const int tile  = blockIdx.x * 4 + wid;
  const int m0    = tile * 16;
  const int kcol  = lane & 15;
  const int khalf = lane >> 4;

  v16h Bt;
  #pragma unroll
  for (int p = 0; p < 8; ++p) {
    int k = 2 * p + 16 * khalf;
    Bt[2 * p]     = (_Float16)W2[k * O_ + kcol];
    Bt[2 * p + 1] = (_Float16)W2[(k + 1) * O_ + kcol];
  }
  v16h A;
  #pragma unroll
  for (int p = 0; p < 8; ++p) {
    int k = kmapA(p, khalf);
    A[2 * p]     = (_Float16)h1[(size_t)(m0 + kcol) * H_ + k];
    A[2 * p + 1] = (_Float16)h1[(size_t)(m0 + kcol) * H_ + k + 1];
  }
  const v8f cz = {0,0,0,0,0,0,0,0};
  v8f C = __builtin_amdgcn_wmma_f32_16x16x32_f16(false, A, false, Bt,
                                                 (short)0, cz, false, false);
  #pragma unroll
  for (int p = 0; p < 8; ++p)
    xw2[(size_t)(m0 + p + 8 * khalf) * O_ + kcol] = C[p];
}

// ---------------------------------------------------------------------------
// Graph-norm + scatter + epilogue kernels
// ---------------------------------------------------------------------------
__global__ void k_fill(float* p, float v, int n) {
  int i = blockIdx.x * blockDim.x + threadIdx.x;
  if (i < n) p[i] = v;
}
__global__ void k_deg(const int* __restrict__ tgt, const float* __restrict__ w,
                      float* deg, int n) {
  int e = blockIdx.x * blockDim.x + threadIdx.x;
  if (e < n) unsafeAtomicAdd(&deg[tgt[e]], w[e]);
}
__global__ void k_dinv(const float* __restrict__ deg, float* dinv, int n) {
  int i = blockIdx.x * blockDim.x + threadIdx.x;
  if (i < n) { float d = deg[i]; dinv[i] = d > 0.0f ? rsqrtf(d) : 0.0f; }
}
__global__ void k_norm(const int* __restrict__ src, const int* __restrict__ tgt,
                       const float* __restrict__ w, const float* __restrict__ dinv,
                       float* __restrict__ nrm, int n) {
  int e = blockIdx.x * blockDim.x + threadIdx.x;
  if (e < n) nrm[e] = dinv[src[e]] * w[e] * dinv[tgt[e]];
}
// one wave per (edge, batch); lane = channel
template <int C>
__global__ __launch_bounds__(256)
void k_scatter(const float* __restrict__ xw, const int* __restrict__ src,
               const int* __restrict__ tgt, const float* __restrict__ nrm,
               float* __restrict__ acc) {
  int wid = threadIdx.x >> 5, lane = threadIdx.x & 31;
  int e = blockIdx.x * 8 + wid;
  int b = blockIdx.y;
  if (e >= E_ || lane >= C) return;
  int s = src[e], tg = tgt[e];
  float v = nrm[e] * xw[((size_t)b * N_ + s) * C + lane];
  unsafeAtomicAdd(&acc[((size_t)b * N_ + tg) * C + lane], v);
}
// h1 = relu(acc1 + dinv^2 * xw1 + b1), in place on acc1
__global__ void k_finish1(float* __restrict__ acc1, const float* __restrict__ xw1,
                          const float* __restrict__ dinv, const float* __restrict__ b1) {
  int idx = blockIdx.x * blockDim.x + threadIdx.x;
  if (idx >= M_ * H_) return;
  int c = idx & (H_ - 1);
  int node = (idx >> 5) % N_;
  float di = dinv[node];
  float v = acc1[idx] + di * di * xw1[idx] + b1[c];
  acc1[idx] = fmaxf(v, 0.0f);
}
// out = ((acc2 + dinv^2*xw2 + b2) @ Wfc) + bfc
__global__ void k_final(const float* __restrict__ acc2, const float* __restrict__ xw2,
                        const float* __restrict__ dinv, const float* __restrict__ b2,
                        const float* __restrict__ Wfc, const float* __restrict__ bfc,
                        float* __restrict__ out) {
  int idx = blockIdx.x * blockDim.x + threadIdx.x;
  if (idx >= M_) return;
  int node = idx % N_;
  float di2 = dinv[node] * dinv[node];
  float s = bfc[0];
  #pragma unroll
  for (int c = 0; c < O_; ++c)
    s += (acc2[(size_t)idx * O_ + c] + di2 * xw2[(size_t)idx * O_ + c] + b2[c]) * Wfc[c];
  out[idx] = s;
}

// ---------------------------------------------------------------------------
extern "C" void kernel_launch(void* const* d_in, const int* in_sizes, int n_in,
                              void* d_out, int out_size, void* d_ws, size_t ws_size,
                              hipStream_t stream) {
  const float* x    = (const float*)d_in[0];
  const int*   ei   = (const int*)  d_in[1];
  const float* ew   = (const float*)d_in[2];
  const float* w_ih = (const float*)d_in[3];
  const float* w_hh = (const float*)d_in[4];
  const float* b_ih = (const float*)d_in[5];
  const float* b_hh = (const float*)d_in[6];
  const float* W1   = (const float*)d_in[7];
  const float* b1   = (const float*)d_in[8];
  const float* W2   = (const float*)d_in[9];
  const float* b2   = (const float*)d_in[10];
  const float* Wfc  = (const float*)d_in[11];
  const float* bfc  = (const float*)d_in[12];
  float* out = (float*)d_out;

  const int* src = ei;
  const int* tgt = ei + E_;

  // workspace layout (floats): ~62.5 MB total
  float* Wp   = (float*)d_ws;
  float* deg  = Wp;                        // N
  float* dinv = Wp + 16384;                // N
  float* nrm  = Wp + 32768;                // E
  float* xw1  = Wp + 262144;               // M*H
  float* acc1 = xw1 + (size_t)M_ * H_;     // M*H (becomes h1)
  float* xw2  = acc1 + (size_t)M_ * H_;    // M*O
  float* acc2 = xw2 + (size_t)M_ * O_;     // M*O

  k_fill<<<dim3((N_ + 255) / 256), 256, 0, stream>>>(deg, 1.0f, N_);       // self-loop weight
  k_fill<<<dim3((M_ * H_ + 255) / 256), 256, 0, stream>>>(acc1, 0.0f, M_ * H_);
  k_fill<<<dim3((M_ * O_ + 255) / 256), 256, 0, stream>>>(acc2, 0.0f, M_ * O_);

  k_deg <<<dim3((E_ + 255) / 256), 256, 0, stream>>>(tgt, ew, deg, E_);
  k_dinv<<<dim3((N_ + 255) / 256), 256, 0, stream>>>(deg, dinv, N_);
  k_norm<<<dim3((E_ + 255) / 256), 256, 0, stream>>>(src, tgt, ew, dinv, nrm, E_);

  k_gru<<<dim3(TILES_ / 4), 128, 0, stream>>>(x, w_ih, w_hh, b_ih, b_hh, W1, xw1);

  k_scatter<H_><<<dim3((E_ + 7) / 8, B_), 256, 0, stream>>>(xw1, src, tgt, nrm, acc1);
  k_finish1<<<dim3((M_ * H_ + 255) / 256), 256, 0, stream>>>(acc1, xw1, dinv, b1);

  k_xw2<<<dim3(TILES_ / 4), 128, 0, stream>>>(acc1, W2, xw2);

  k_scatter<O_><<<dim3((E_ + 7) / 8, B_), 256, 0, stream>>>(xw2, src, tgt, nrm, acc2);
  k_final<<<dim3((M_ + 255) / 256), 256, 0, stream>>>(acc2, xw2, dinv, b2, Wfc, bfc, out);
}
